// MultiHeadAttentionXL_3204045603388
// MI455X (gfx1250) — matrix operand order repaired and checked by reading, not compiled, not run on hardware
//
#include <hip/hip_runtime.h>

// ---------------------------------------------------------------------------
// MultiHeadAttention-XL for MI455X (gfx1250, wave32, WMMA bf16)
//   X:[16,512,2048] f32, Wq/Wk/Wv/Wo:[2048,2048] f32, rel_pos:[1023,64] f32
//   out = softmax(QK^T/8 + relbias) V  -> @ Wo^T, f32 [16,512,2048]
// bf16 downconvert once; all GEMMs via v_wmma_f32_16x16x32_bf16; LDS staging
// via double-buffered GLOBAL_LOAD_ASYNC_TO_LDS_B128 (ASYNCcnt pipelining);
// flash attention with K row-major + V transposed in LDS.
// ---------------------------------------------------------------------------

#define HIDDEN   2048
#define HEADS    32
#define HEAD_DIM 64
#define SEQ      512
#define BATCH    16
#define MROWS    (BATCH * SEQ)   // 8192
#define TABLE    1023
#define MAXPOS   512

typedef __attribute__((ext_vector_type(16))) __bf16 v16bf;
typedef __attribute__((ext_vector_type(8)))  float  v8f;

__device__ __forceinline__ v16bf mk_frag(uint4 a, uint4 b) {
    union { uint4 u[2]; v16bf v; } t;
    t.u[0] = a; t.u[1] = b;
    return t.v;
}

__device__ __forceinline__ unsigned short f2bf(float f) {
    unsigned int u = __float_as_uint(f);
    u += 0x7FFFu + ((u >> 16) & 1u);   // round-to-nearest-even
    return (unsigned short)(u >> 16);
}

// Low 32 bits of a generic pointer into LDS == LDS byte offset (flat-LDS
// aperture mapping, ISA 10.2): usable as the VDST operand of async-to-LDS.
__device__ __forceinline__ unsigned lds_off(const void* p) {
    return (unsigned)(unsigned long long)p;
}

// async copy of 16B from global to LDS, tracked by ASYNCcnt (no VGPR data)
__device__ __forceinline__ void async_b128(unsigned ldsoff, const void* gp) {
    unsigned long long ga = (unsigned long long)gp;
    asm volatile("global_load_async_to_lds_b128 %0, %1, off"
                 :: "v"(ldsoff), "v"(ga) : "memory");
}
__device__ __forceinline__ void wait_async0() {
    asm volatile("s_wait_asynccnt 0" ::: "memory");
}
// drain everything except the 8 in-flight copies of the *next* tile
__device__ __forceinline__ void wait_async8() {
    asm volatile("s_wait_asynccnt 8" ::: "memory");
}

// ---------------------------------------------------------------- convert ---
__global__ void cvt_f32_bf16(const float* __restrict__ in,
                             unsigned short* __restrict__ out, int n) {
    int i = blockIdx.x * blockDim.x + threadIdx.x;
    int stride = gridDim.x * blockDim.x;
    for (; i < n; i += stride) out[i] = f2bf(in[i]);
}

// --------------------------------------------------------------- rel bias ---
// bias[q,c] = mean_d rel_pos[clip(c-q+511, 0, 1022)][d]
__global__ void rel_bias_kernel(const float* __restrict__ rel,
                                float* __restrict__ bias) {
    int q = blockIdx.x;
    for (int c = threadIdx.x; c < SEQ; c += blockDim.x) {
        int r = c - q + (MAXPOS - 1);
        r = min(max(r, 0), TABLE - 1);
        const float4* p = (const float4*)(rel + (size_t)r * HEAD_DIM);
        float s = 0.f;
#pragma unroll
        for (int i = 0; i < 16; i++) {
            float4 v = p[i];
            s += v.x + v.y + v.z + v.w;
        }
        bias[(size_t)q * SEQ + c] = s * (1.0f / HEAD_DIM);
    }
}

// ------------------------------------------------------------------- GEMM ---
// C[M=8192, N=2048] = A[M,2048] @ Bw[N,2048]^T   (A, Bw bf16 row-major)
// Block tile 128x128, 8 waves (2x4), wave tile 64x32, K-step 64,
// double-buffered async-to-LDS staging (ping-pong, ASYNCcnt-gated).
// MODE 0: write bf16, scattered to [B,HEADS,S,HEAD_DIM]
// MODE 1: write f32 row-major [M,N]
#define GTILE (128 * 72)          // one 128x64 bf16 tile, stride 72
template <int MODE>
__global__ __launch_bounds__(256)
void gemm_bf16(const unsigned short* __restrict__ A,
               const unsigned short* __restrict__ Bw,
               void* __restrict__ Out) {
    extern __shared__ unsigned short sm[];   // [2 bufs][A tile | B tile]

    const int tid  = threadIdx.x;
    const int lane = tid & 31;
    const int wave = tid >> 5;
    const int mw   = wave >> 2;   // 0..1  -> 64-row wave slab
    const int nw   = wave & 3;    // 0..3  -> 32-col wave slab
    const int rowBase = blockIdx.y * 128;
    const int colBase = blockIdx.x * 128;

    const int lrow  = lane & 15;
    const int lhalf = lane >> 4;
    const int klo   = lhalf * 8;    // A-frag chunk base (ISA 16-bit A layout)
    const int khalf = lhalf * 8;    // B-frag chunk base (same k order as A)

    // per-thread chunk of the cooperative tile fill (4 x 16B per matrix)
    auto stage = [&](int k0, int buf) {
        unsigned short* As = sm + buf * (2 * GTILE);
        unsigned short* Bs = As + GTILE;
#pragma unroll
        for (int i = 0; i < 4; i++) {
            int c = tid + i * 256;
            int r = c >> 3, cc = c & 7;
            async_b128(lds_off(As + r * 72 + cc * 8),
                       A  + (size_t)(rowBase + r) * HIDDEN + k0 + cc * 8);
            async_b128(lds_off(Bs + r * 72 + cc * 8),
                       Bw + (size_t)(colBase + r) * HIDDEN + k0 + cc * 8);
        }
    };

    v8f zero8 = {0.f, 0.f, 0.f, 0.f, 0.f, 0.f, 0.f, 0.f};
    v8f acc[4][2];
#pragma unroll
    for (int i = 0; i < 4; i++)
#pragma unroll
        for (int j = 0; j < 2; j++) acc[i][j] = zero8;

    stage(0, 0);                               // prologue fill of buffer 0
    const int KTILES = HIDDEN / 64;            // 32
    for (int kt = 0; kt < KTILES; kt++) {
        const int buf = kt & 1;
        if (kt + 1 < KTILES) {
            stage((kt + 1) * 64, buf ^ 1);     // overlap next-tile fill
            wait_async8();                     // current tile done; next in flight
        } else {
            wait_async0();
        }
        __syncthreads();

        const unsigned short* As = sm + buf * (2 * GTILE);
        const unsigned short* Bs = As + GTILE;
#pragma unroll
        for (int kc = 0; kc < 2; kc++) {       // two 32-wide K chunks
            v16bf af[4], bfr[2];
#pragma unroll
            for (int mt = 0; mt < 4; mt++) {
                const unsigned short* p =
                    As + (mw * 64 + mt * 16 + lrow) * 72 + kc * 32;
                af[mt] = mk_frag(*(const uint4*)(p + klo),
                                 *(const uint4*)(p + klo + 16));
            }
#pragma unroll
            for (int nt = 0; nt < 2; nt++) {
                const unsigned short* p =
                    Bs + (nw * 32 + nt * 16 + lrow) * 72 + kc * 32 + khalf;
                bfr[nt] = mk_frag(*(const uint4*)(p),
                                  *(const uint4*)(p + 16));
            }
#pragma unroll
            for (int mt = 0; mt < 4; mt++)
#pragma unroll
                for (int nt = 0; nt < 2; nt++)
                    acc[mt][nt] = __builtin_amdgcn_wmma_f32_16x16x32_bf16(
                        false, af[mt], false, bfr[nt], (short)0, acc[mt][nt],
                        false, false);
        }
        __syncthreads();   // guard buffer reuse before next overwrite
    }

    // epilogue (C layout: lanes 0-15 -> rows v, lanes 16-31 -> rows v+8)
#pragma unroll
    for (int mt = 0; mt < 4; mt++) {
#pragma unroll
        for (int nt = 0; nt < 2; nt++) {
#pragma unroll
            for (int v = 0; v < 8; v++) {
                int r = rowBase + mw * 64 + mt * 16 + v + lhalf * 8;
                int n = colBase + nw * 32 + nt * 16 + lrow;
                float val = acc[mt][nt][v];
                if (MODE == 0) {
                    int b = r >> 9, s = r & 511, h = n >> 6, d = n & 63;
                    ((unsigned short*)Out)[(((size_t)(b * HEADS + h) * SEQ + s) * HEAD_DIM) + d]
                        = f2bf(val);
                } else {
                    ((float*)Out)[(size_t)r * HIDDEN + n] = val;
                }
            }
        }
    }
}

// -------------------------------------------------------------- attention ---
// grid (B*HEADS, SEQ/128); 8 waves, each owns 16 query rows; flash softmax.
__global__ __launch_bounds__(256)
void attn_kernel(const unsigned short* __restrict__ Qb,
                 const unsigned short* __restrict__ Kb,
                 const unsigned short* __restrict__ Vb,
                 const float* __restrict__ bias,
                 unsigned short* __restrict__ ctx) {
    extern __shared__ char smem[];
    unsigned short* Ks = (unsigned short*)smem;   // [512][72] row-major K
    unsigned short* Vt = Ks + 512 * 72;           // [64][520] transposed V
    unsigned short* Pb = Vt + 64 * 520;           // per-wave [16][40] P tile

    const int tid = threadIdx.x, lane = tid & 31, wave = tid >> 5;
    const int bh = blockIdx.x;
    const int b = bh >> 5, h = bh & 31;
    const int sBase = blockIdx.y * 128;
    const int lrow = lane & 15, lhalf = lane >> 4;

    const size_t headOff = (size_t)bh * SEQ * HEAD_DIM;
    const unsigned short* Kg = Kb + headOff;
    const unsigned short* Vg = Vb + headOff;
    const unsigned short* Qg = Qb + headOff + (size_t)sBase * HEAD_DIM;

    // K: async copy straight into LDS (row-major, stride 72), overlapped
    // with the manual V load + transpose scatter into Vt.
#pragma unroll
    for (int i = 0; i < 16; i++) {
        int c = tid + i * 256;          // 4096 x 16B chunks
        int r = c >> 3, cc = c & 7;
        async_b128(lds_off(Ks + r * 72 + cc * 8),
                   Kg + (size_t)r * HEAD_DIM + cc * 8);
        uint4 vv = *(const uint4*)(Vg + (size_t)r * HEAD_DIM + cc * 8);
        union { uint4 u; unsigned short e[8]; } tv;
        tv.u = vv;
#pragma unroll
        for (int e = 0; e < 8; e++)
            Vt[(cc * 8 + e) * 520 + r] = tv.e[e];
    }
    wait_async0();
    __syncthreads();

    // Q A-fragments for this wave's 16 rows (dims 0..31 and 32..63)
    v16bf qa[2];
    {
        const unsigned short* qrow = Qg + (size_t)(wave * 16 + lrow) * HEAD_DIM;
        const int aklo = lhalf * 8;
#pragma unroll
        for (int f = 0; f < 2; f++)
            qa[f] = mk_frag(*(const uint4*)(qrow + f * 32 + aklo),
                            *(const uint4*)(qrow + f * 32 + aklo + 16));
    }

    v8f zero8 = {0.f, 0.f, 0.f, 0.f, 0.f, 0.f, 0.f, 0.f};
    v8f O[4] = {zero8, zero8, zero8, zero8};
    float mi[8], li[8];
#pragma unroll
    for (int v = 0; v < 8; v++) { mi[v] = -INFINITY; li[v] = 0.f; }

    unsigned short* Pw = Pb + wave * 16 * 40;
    const float scale = 0.125f;              // 1/sqrt(64)
    const int qRow0 = sBase + wave * 16;     // sequence-local query base

    for (int j = 0; j < SEQ; j += 32) {
        // scores: two 16x16 tiles, K-dim 64 = 2 wmmas each
        v8f s[2];
#pragma unroll
        for (int t = 0; t < 2; t++) {
            v8f z = zero8;
#pragma unroll
            for (int f = 0; f < 2; f++) {
                const unsigned short* kp =
                    Ks + (j + t * 16 + lrow) * 72 + f * 32 + lhalf * 8;
                v16bf kb = mk_frag(*(const uint4*)(kp),
                                   *(const uint4*)(kp + 16));
                z = __builtin_amdgcn_wmma_f32_16x16x32_bf16(
                    false, qa[f], false, kb, (short)0, z, false, false);
            }
            s[t] = z;
        }
        // scale + relative position bias
#pragma unroll
        for (int t = 0; t < 2; t++)
#pragma unroll
            for (int v = 0; v < 8; v++) {
                int qr = qRow0 + v + lhalf * 8;
                int kc = j + t * 16 + lrow;
                s[t][v] = s[t][v] * scale + bias[(size_t)qr * SEQ + kc];
            }
        // flash softmax update (row stats live per-VGPR; width-16 reductions)
        float alpha[8];
#pragma unroll
        for (int v = 0; v < 8; v++) {
            float x = fmaxf(s[0][v], s[1][v]);
#pragma unroll
            for (int m = 8; m >= 1; m >>= 1) x = fmaxf(x, __shfl_xor(x, m, 16));
            float mn = fmaxf(mi[v], x);
            alpha[v] = __expf(mi[v] - mn);
            mi[v] = mn;
            s[0][v] = __expf(s[0][v] - mn);
            s[1][v] = __expf(s[1][v] - mn);
            float rs = s[0][v] + s[1][v];
#pragma unroll
            for (int m = 8; m >= 1; m >>= 1) rs += __shfl_xor(rs, m, 16);
            li[v] = li[v] * alpha[v] + rs;
        }
#pragma unroll
        for (int n = 0; n < 4; n++)
#pragma unroll
            for (int v = 0; v < 8; v++) O[n][v] *= alpha[v];

        // relayout P (C layout -> A layout) via per-wave LDS round-trip
#pragma unroll
        for (int t = 0; t < 2; t++)
#pragma unroll
            for (int v = 0; v < 8; v++)
                Pw[(v + lhalf * 8) * 40 + t * 16 + lrow] = f2bf(s[t][v]);
        asm volatile("s_wait_dscnt 0" ::: "memory");

        const unsigned short* pr = Pw + lrow * 40;
        const int aklo = lhalf * 8;
        v16bf pa = mk_frag(*(const uint4*)(pr + aklo),
                           *(const uint4*)(pr + aklo + 16));

        // O += P @ V  (V transposed in LDS -> contiguous B fragments)
#pragma unroll
        for (int n = 0; n < 4; n++) {
            const unsigned short* vp =
                Vt + (n * 16 + lrow) * 520 + j + lhalf * 8;
            v16bf vb = mk_frag(*(const uint4*)(vp),
                               *(const uint4*)(vp + 16));
            O[n] = __builtin_amdgcn_wmma_f32_16x16x32_bf16(
                false, pa, false, vb, (short)0, O[n], false, false);
        }
    }

    // normalize and write ctx (bf16, [B*S, HIDDEN] row-major)
#pragma unroll
    for (int n = 0; n < 4; n++)
#pragma unroll
        for (int v = 0; v < 8; v++) {
            int r = b * SEQ + qRow0 + v + lhalf * 8;
            int cc = h * HEAD_DIM + n * 16 + lrow;
            ctx[(size_t)r * HIDDEN + cc] = f2bf(O[n][v] / li[v]);
        }
}

// ----------------------------------------------------------------- launch ---
extern "C" void kernel_launch(void* const* d_in, const int* in_sizes, int n_in,
                              void* d_out, int out_size, void* d_ws,
                              size_t ws_size, hipStream_t stream) {
    (void)in_sizes; (void)n_in; (void)out_size; (void)ws_size;

    const float* X   = (const float*)d_in[0];
    const float* wq  = (const float*)d_in[1];
    const float* wk  = (const float*)d_in[2];
    const float* wv  = (const float*)d_in[3];
    const float* wo  = (const float*)d_in[4];
    const float* rel = (const float*)d_in[5];

    char* ws = (char*)d_ws;
    size_t off = 0;
    auto carve = [&](size_t bytes) {
        void* p = ws + off;
        off += (bytes + 255) & ~(size_t)255;
        return p;
    };
    unsigned short* Xb  = (unsigned short*)carve((size_t)MROWS * HIDDEN * 2);
    unsigned short* Wqb = (unsigned short*)carve((size_t)HIDDEN * HIDDEN * 2);
    unsigned short* Wkb = (unsigned short*)carve((size_t)HIDDEN * HIDDEN * 2);
    unsigned short* Wvb = (unsigned short*)carve((size_t)HIDDEN * HIDDEN * 2);
    unsigned short* Wob = (unsigned short*)carve((size_t)HIDDEN * HIDDEN * 2);
    unsigned short* Qb  = (unsigned short*)carve((size_t)MROWS * HIDDEN * 2);
    unsigned short* Kb  = (unsigned short*)carve((size_t)MROWS * HIDDEN * 2);
    unsigned short* Vb  = (unsigned short*)carve((size_t)MROWS * HIDDEN * 2);
    unsigned short* Ctx = (unsigned short*)carve((size_t)MROWS * HIDDEN * 2);
    float*          Bias = (float*)carve((size_t)SEQ * SEQ * 4);

    // 1) downconvert to bf16
    cvt_f32_bf16<<<2048, 256, 0, stream>>>(X,  Xb,  MROWS * HIDDEN);
    cvt_f32_bf16<<<1024, 256, 0, stream>>>(wq, Wqb, HIDDEN * HIDDEN);
    cvt_f32_bf16<<<1024, 256, 0, stream>>>(wk, Wkb, HIDDEN * HIDDEN);
    cvt_f32_bf16<<<1024, 256, 0, stream>>>(wv, Wvb, HIDDEN * HIDDEN);
    cvt_f32_bf16<<<1024, 256, 0, stream>>>(wo, Wob, HIDDEN * HIDDEN);

    // 2) relative-position bias table
    rel_bias_kernel<<<SEQ, 256, 0, stream>>>(rel, Bias);

    // 3) QKV projections (WMMA GEMMs), scattered to [B,H,S,D] bf16
    dim3 ggrid(HIDDEN / 128, MROWS / 128);
    size_t gsmem = (size_t)2 * 2 * GTILE * sizeof(unsigned short);   // 72 KB
    gemm_bf16<0><<<ggrid, 256, gsmem, stream>>>(Xb, Wqb, Qb);
    gemm_bf16<0><<<ggrid, 256, gsmem, stream>>>(Xb, Wkb, Kb);
    gemm_bf16<0><<<ggrid, 256, gsmem, stream>>>(Xb, Wvb, Vb);

    // 4) flash attention with relative bias
    size_t smem = (size_t)(512 * 72 + 64 * 520 + 8 * 16 * 40) * sizeof(unsigned short);
    attn_kernel<<<dim3(BATCH * HEADS, SEQ / 128), 256, smem, stream>>>(
        Qb, Kb, Vb, Bias, Ctx);

    // 5) output projection -> f32 d_out
    gemm_bf16<1><<<ggrid, 256, gsmem, stream>>>(Ctx, Wob, (float*)d_out);
}
